// scGAT_with_BN_40106404610224
// MI455X (gfx1250) — compile-verified, matrix operand order checked by reading, not run on hardware
//
#include <hip/hip_runtime.h>
#include <hip/hip_bf16.h>

// ---------- types for CDNA5 WMMA ----------
typedef __attribute__((ext_vector_type(16))) __bf16 v16bf;
typedef __attribute__((ext_vector_type(8)))  float  v8f;

#define DIN   1000
#define KPAD  1024
#define HEADS 8
#define C1    8
#define D2IN  64      // H*C1
#define DOUT  20
#define H2OUT 160     // H*DOUT

// ---------- helpers: order-preserving float<->uint for atomicMax ----------
__device__ __forceinline__ unsigned fkey(float f) {
    int i = __float_as_int(f);
    return (i >= 0) ? ((unsigned)i | 0x80000000u) : ~(unsigned)i;
}
__device__ __forceinline__ float fval(unsigned k) {
    int i = (k & 0x80000000u) ? (int)(k & 0x7FFFFFFFu) : (int)(~k);
    return __int_as_float(i);
}

// ---------- BatchNorm statistics ----------
__global__ void bn_stats_kernel(const float* __restrict__ x, float* __restrict__ sums,
                                float* __restrict__ sumsq, int N, int rowsPer) {
    int j = blockIdx.x * blockDim.x + threadIdx.x;
    if (j >= DIN) return;
    int r0 = blockIdx.y * rowsPer;
    int r1 = min(r0 + rowsPer, N);
    float s = 0.f, q = 0.f;
    for (int r = r0; r < r1; ++r) {
        float v = x[(size_t)r * DIN + j];
        s += v; q += v * v;
    }
    atomicAdd(&sums[j], s);
    atomicAdd(&sumsq[j], q);
}

__global__ void bn_finalize_kernel(const float* __restrict__ sums, const float* __restrict__ sumsq,
                                   const float* __restrict__ gamma, const float* __restrict__ beta,
                                   float* __restrict__ scale, float* __restrict__ shift, int N) {
    int j = blockIdx.x * blockDim.x + threadIdx.x;
    if (j >= DIN) return;
    float inv = 1.f / (float)N;
    float mu  = sums[j] * inv;
    float var = sumsq[j] * inv - mu * mu;
    float sc  = gamma[j] * rsqrtf(var + 1e-5f);
    scale[j] = sc;
    shift[j] = beta[j] - mu * sc;
}

// ---------- pre-swizzle weights into WMMA B-operand lane layout ----------
// out linear index = ((ks*tiles + tile)*32 + lane)*16 + e
__global__ void swizzle_w_kernel(const float* __restrict__ W, __bf16* __restrict__ out,
                                 int ksteps, int tiles, int Krows, int Ncols) {
    int idx = blockIdx.x * blockDim.x + threadIdx.x;
    int total = ksteps * tiles * 512;
    if (idx >= total) return;
    int e    = idx & 15;
    int lane = (idx >> 4) & 31;
    int tile = (idx >> 9) % tiles;
    int ks   = (idx >> 9) / tiles;
    int K    = (ks << 5) + ((e < 8) ? e : e + 8) + ((lane >= 16) ? 8 : 0);
    int col  = (tile << 4) + (lane & 15);
    float v  = (K < Krows && col < Ncols) ? W[(size_t)K * Ncols + col] : 0.f;
    out[idx] = (__bf16)v;
}

// ---------- GEMM1: h1[N,64] = BN(x)[N,1000] @ W1 (bf16 WMMA, fp32 acc) ----------
__global__ __launch_bounds__(256)
void gemm1_kernel(const float* __restrict__ x, const float* __restrict__ scale,
                  const float* __restrict__ shift, const __bf16* __restrict__ w1s,
                  float* __restrict__ h1, int nTiles) {
    __shared__ float s_sc[KPAD];
    __shared__ float s_sh[KPAD];
    for (int i = threadIdx.x; i < KPAD; i += blockDim.x) {
        s_sc[i] = (i < DIN) ? scale[i] : 0.f;
        s_sh[i] = (i < DIN) ? shift[i] : 0.f;
    }
    __syncthreads();
    int wave = threadIdx.x >> 5;
    int lane = threadIdx.x & 31;
    int t = blockIdx.x * 8 + wave;
    if (t >= nTiles) return;                 // wave-uniform -> EXEC stays all-1s
    int row0 = t << 4;
    int m    = lane & 15;
    int kh   = (lane >> 4) << 3;             // 0 or 8
    const float* xr = x + (size_t)(row0 + m) * DIN;
    v8f acc0 = {}, acc1 = {}, acc2 = {}, acc3 = {};
    for (int ks = 0; ks < 32; ++ks) {
        int kb = ks << 5;
        v16bf a;
#pragma unroll
        for (int e = 0; e < 16; ++e) {
            int K  = kb + ((e < 8) ? e : e + 8) + kh;
            int Kc = (K < DIN) ? K : (DIN - 1);          // safe read; s_sc[K>=DIN]==0
            float v = xr[Kc] * s_sc[K] + s_sh[K];
            a[e] = (__bf16)v;
        }
        const __bf16* bb = w1s + (size_t)(ks * 4) * 512 + (size_t)lane * 16;
        v16bf b0 = *(const v16bf*)(bb);
        v16bf b1 = *(const v16bf*)(bb + 512);
        v16bf b2 = *(const v16bf*)(bb + 1024);
        v16bf b3 = *(const v16bf*)(bb + 1536);
        acc0 = __builtin_amdgcn_wmma_f32_16x16x32_bf16(false, a, false, b0, (short)0, acc0, false, false);
        acc1 = __builtin_amdgcn_wmma_f32_16x16x32_bf16(false, a, false, b1, (short)0, acc1, false, false);
        acc2 = __builtin_amdgcn_wmma_f32_16x16x32_bf16(false, a, false, b2, (short)0, acc2, false, false);
        acc3 = __builtin_amdgcn_wmma_f32_16x16x32_bf16(false, a, false, b3, (short)0, acc3, false, false);
    }
    int hi  = (lane >> 4) << 3;
    int col = lane & 15;
#pragma unroll
    for (int v = 0; v < 8; ++v) {
        size_t r = (size_t)(row0 + v + hi) * D2IN;
        h1[r +      col] = acc0[v];
        h1[r + 16 + col] = acc1[v];
        h1[r + 32 + col] = acc2[v];
        h1[r + 48 + col] = acc3[v];
    }
}

// ---------- GEMM2: h2[N,160] = elu_out[N,64] @ W2 ----------
__global__ __launch_bounds__(256)
void gemm2_kernel(const float* __restrict__ in, const __bf16* __restrict__ w2s,
                  float* __restrict__ h2, int nTiles) {
    int wave = threadIdx.x >> 5;
    int lane = threadIdx.x & 31;
    int t = blockIdx.x * 8 + wave;
    if (t >= nTiles) return;
    int row0 = t << 4;
    int m    = lane & 15;
    int kh   = (lane >> 4) << 3;
    const float* ir = in + (size_t)(row0 + m) * D2IN;
    v8f acc[10];
#pragma unroll
    for (int i = 0; i < 10; ++i) acc[i] = (v8f){};
    for (int ks = 0; ks < 2; ++ks) {
        int kb = ks << 5;
        v16bf a;
#pragma unroll
        for (int e = 0; e < 16; ++e) {
            int K = kb + ((e < 8) ? e : e + 8) + kh;     // K < 64 always
            a[e] = (__bf16)ir[K];
        }
#pragma unroll
        for (int tl = 0; tl < 10; ++tl) {
            v16bf b = *(const v16bf*)(w2s + (size_t)((ks * 10 + tl) * 32 + lane) * 16);
            acc[tl] = __builtin_amdgcn_wmma_f32_16x16x32_bf16(false, a, false, b, (short)0, acc[tl], false, false);
        }
    }
    int hi  = (lane >> 4) << 3;
    int col = lane & 15;
#pragma unroll
    for (int v = 0; v < 8; ++v) {
        size_t r = (size_t)(row0 + v + hi) * H2OUT;
#pragma unroll
        for (int tl = 0; tl < 10; ++tl) h2[r + tl * 16 + col] = acc[tl][v];
    }
}

// ---------- per-node attention coefficients: al = sum_c h[n,h,c]*a[h,c] ----------
__global__ void attn_coef_kernel(const float* __restrict__ h, const float* __restrict__ a_src,
                                 const float* __restrict__ a_dst, float* __restrict__ al_s,
                                 float* __restrict__ al_d, int N, int C, int stride) {
    int idx = blockIdx.x * blockDim.x + threadIdx.x;
    if (idx >= N * HEADS) return;
    int n  = idx >> 3;
    int hh = idx & 7;
    const float* hp = h + (size_t)n * stride + hh * C;
    const float* as = a_src + hh * C;
    const float* ad = a_dst + hh * C;
    float ss = 0.f, sd = 0.f;
    for (int c = 0; c < C; ++c) { float v = hp[c]; ss += v * as[c]; sd += v * ad[c]; }
    al_s[idx] = ss;
    al_d[idx] = sd;
}

__device__ __forceinline__ void edge_ids(const int* __restrict__ ei, int eid, int E,
                                         int& src, int& dst) {
    if (eid < E) { src = ei[eid]; dst = ei[E + eid]; }
    else         { src = eid - E; dst = eid - E; }     // self loops
}

// ---------- edge pass 1: segment max ----------
__global__ void edge_max_kernel(const int* __restrict__ ei, int E, int N,
                                const float* __restrict__ al_s, const float* __restrict__ al_d,
                                unsigned* __restrict__ mx) {
    int idx = blockIdx.x * blockDim.x + threadIdx.x;
    if (idx >= (E + N) * HEADS) return;
    int eid = idx >> 3, hh = idx & 7, src, dst;
    edge_ids(ei, eid, E, src, dst);
    float e = al_s[src * HEADS + hh] + al_d[dst * HEADS + hh];
    e = (e >= 0.f) ? e : 0.2f * e;
    atomicMax(&mx[dst * HEADS + hh], fkey(e));
}

// ---------- edge pass 2: segment denom ----------
__global__ void edge_den_kernel(const int* __restrict__ ei, int E, int N,
                                const float* __restrict__ al_s, const float* __restrict__ al_d,
                                const unsigned* __restrict__ mx, float* __restrict__ den) {
    int idx = blockIdx.x * blockDim.x + threadIdx.x;
    if (idx >= (E + N) * HEADS) return;
    int eid = idx >> 3, hh = idx & 7, src, dst;
    edge_ids(ei, eid, E, src, dst);
    float e = al_s[src * HEADS + hh] + al_d[dst * HEADS + hh];
    e = (e >= 0.f) ? e : 0.2f * e;
    float ex = __expf(e - fval(mx[dst * HEADS + hh]));
    atomicAdd(&den[dst * HEADS + hh], ex);
}

// ---------- edge pass 3: weighted scatter-add of messages ----------
__global__ void edge_msg_kernel(const int* __restrict__ ei, int E, int N,
                                const float* __restrict__ al_s, const float* __restrict__ al_d,
                                const unsigned* __restrict__ mx, const float* __restrict__ den,
                                const float* __restrict__ hfeat, float* __restrict__ outacc,
                                int C, int hStride, int oStride, int oHeadStride, float msc) {
    int idx = blockIdx.x * blockDim.x + threadIdx.x;
    if (idx >= (E + N) * HEADS) return;
    int eid = idx >> 3, hh = idx & 7, src, dst;
    edge_ids(ei, eid, E, src, dst);
    float e = al_s[src * HEADS + hh] + al_d[dst * HEADS + hh];
    e = (e >= 0.f) ? e : 0.2f * e;
    int di = dst * HEADS + hh;
    float alpha = __expf(e - fval(mx[di])) / (den[di] + 1e-16f) * msc;
    const float* hp = hfeat + (size_t)src * hStride + hh * C;
    float* op = outacc + (size_t)dst * oStride + hh * oHeadStride;
#pragma unroll 4
    for (int c = 0; c < C; ++c) atomicAdd(&op[c], alpha * hp[c]);
}

// ---------- bias + ELU on layer-1 output ----------
__global__ void bias_elu_kernel(float* __restrict__ o, const float* __restrict__ b, int total) {
    int idx = blockIdx.x * blockDim.x + threadIdx.x;
    if (idx >= total) return;
    float v = o[idx] + b[idx & (D2IN - 1)];
    o[idx] = (v > 0.f) ? v : (__expf(v) - 1.f);
}

// ---------- final: + b2, log_softmax ----------
__global__ void final_kernel(const float* __restrict__ acc, const float* __restrict__ b2,
                             float* __restrict__ out, int N) {
    int n = blockIdx.x * blockDim.x + threadIdx.x;
    if (n >= N) return;
    float v[DOUT];
    float m = -3.4e38f;
#pragma unroll
    for (int c = 0; c < DOUT; ++c) { v[c] = acc[(size_t)n * DOUT + c] + b2[c]; m = fmaxf(m, v[c]); }
    float s = 0.f;
#pragma unroll
    for (int c = 0; c < DOUT; ++c) s += __expf(v[c] - m);
    float l = m + logf(s);
#pragma unroll
    for (int c = 0; c < DOUT; ++c) out[(size_t)n * DOUT + c] = v[c] - l;
}

extern "C" void kernel_launch(void* const* d_in, const int* in_sizes, int n_in,
                              void* d_out, int out_size, void* d_ws, size_t ws_size,
                              hipStream_t stream) {
    const float* x      = (const float*)d_in[0];
    const int*   ei     = (const int*)  d_in[1];
    const float* gamma  = (const float*)d_in[2];
    const float* beta   = (const float*)d_in[3];
    const float* W1     = (const float*)d_in[4];
    const float* a_src1 = (const float*)d_in[5];
    const float* a_dst1 = (const float*)d_in[6];
    const float* b1     = (const float*)d_in[7];
    const float* W2     = (const float*)d_in[8];
    const float* a_src2 = (const float*)d_in[9];
    const float* a_dst2 = (const float*)d_in[10];
    const float* b2     = (const float*)d_in[11];
    float* out = (float*)d_out;

    const int N = in_sizes[0] / DIN;
    const int E = in_sizes[1] / 2;

    // ---- workspace layout (256B aligned) ----
    char* w = (char*)d_ws;
    size_t off = 0;
    auto alloc = [&](size_t bytes) -> void* {
        void* p = w + off;
        off += (bytes + 255) & ~(size_t)255;
        return p;
    };
    float*    scale  = (float*)   alloc(DIN * 4);
    float*    shift  = (float*)   alloc(DIN * 4);
    float*    sums   = (float*)   alloc(DIN * 4);
    float*    sumsq  = (float*)   alloc(DIN * 4);
    __bf16*   w1s    = (__bf16*)  alloc((size_t)32 * 4  * 512 * 2);
    __bf16*   w2s    = (__bf16*)  alloc((size_t)2  * 10 * 512 * 2);
    float*    h1     = (float*)   alloc((size_t)N * D2IN * 4);
    float*    al_s1  = (float*)   alloc((size_t)N * HEADS * 4);
    float*    al_d1  = (float*)   alloc((size_t)N * HEADS * 4);
    unsigned* max1   = (unsigned*)alloc((size_t)N * HEADS * 4);
    float*    den1   = (float*)   alloc((size_t)N * HEADS * 4);
    float*    out1   = (float*)   alloc((size_t)N * D2IN * 4);
    float*    h2     = (float*)   alloc((size_t)N * H2OUT * 4);
    float*    al_s2  = (float*)   alloc((size_t)N * HEADS * 4);
    float*    al_d2  = (float*)   alloc((size_t)N * HEADS * 4);
    unsigned* max2   = (unsigned*)alloc((size_t)N * HEADS * 4);
    float*    den2   = (float*)   alloc((size_t)N * HEADS * 4);
    float*    out2   = (float*)   alloc((size_t)N * DOUT * 4);
    (void)ws_size; (void)n_in; (void)out_size;

    // ---- zero the accumulators (capture-legal memsets) ----
    hipMemsetAsync(sums,  0, DIN * 4, stream);
    hipMemsetAsync(sumsq, 0, DIN * 4, stream);
    hipMemsetAsync(max1,  0, (size_t)N * HEADS * 4, stream);   // key 0 == -inf sentinel
    hipMemsetAsync(den1,  0, (size_t)N * HEADS * 4, stream);
    hipMemsetAsync(out1,  0, (size_t)N * D2IN * 4, stream);
    hipMemsetAsync(max2,  0, (size_t)N * HEADS * 4, stream);
    hipMemsetAsync(den2,  0, (size_t)N * HEADS * 4, stream);
    hipMemsetAsync(out2,  0, (size_t)N * DOUT * 4, stream);

    const int rowsPer = 512;
    const int nChunks = (N + rowsPer - 1) / rowsPer;
    bn_stats_kernel<<<dim3((DIN + 255) / 256, nChunks), 256, 0, stream>>>(x, sums, sumsq, N, rowsPer);
    bn_finalize_kernel<<<(DIN + 255) / 256, 256, 0, stream>>>(sums, sumsq, gamma, beta, scale, shift, N);

    swizzle_w_kernel<<<(32 * 4 * 512 + 255) / 256, 256, 0, stream>>>(W1, w1s, 32, 4, DIN, D2IN);
    swizzle_w_kernel<<<(2 * 10 * 512 + 255) / 256, 256, 0, stream>>>(W2, w2s, 2, 10, D2IN, H2OUT);

    const int nTiles = (N + 15) / 16;
    gemm1_kernel<<<(nTiles + 7) / 8, 256, 0, stream>>>(x, scale, shift, w1s, h1, nTiles);

    attn_coef_kernel<<<(N * HEADS + 255) / 256, 256, 0, stream>>>(h1, a_src1, a_dst1, al_s1, al_d1, N, C1, D2IN);

    const int nedge = (E + N) * HEADS;
    const int eb = (nedge + 255) / 256;
    edge_max_kernel<<<eb, 256, 0, stream>>>(ei, E, N, al_s1, al_d1, max1);
    edge_den_kernel<<<eb, 256, 0, stream>>>(ei, E, N, al_s1, al_d1, max1, den1);
    edge_msg_kernel<<<eb, 256, 0, stream>>>(ei, E, N, al_s1, al_d1, max1, den1,
                                            h1, out1, C1, D2IN, D2IN, C1, 1.0f);

    bias_elu_kernel<<<(N * D2IN + 255) / 256, 256, 0, stream>>>(out1, b1, N * D2IN);

    gemm2_kernel<<<(nTiles + 7) / 8, 256, 0, stream>>>(out1, w2s, h2, nTiles);

    attn_coef_kernel<<<(N * HEADS + 255) / 256, 256, 0, stream>>>(h2, a_src2, a_dst2, al_s2, al_d2, N, DOUT, H2OUT);

    edge_max_kernel<<<eb, 256, 0, stream>>>(ei, E, N, al_s2, al_d2, max2);
    edge_den_kernel<<<eb, 256, 0, stream>>>(ei, E, N, al_s2, al_d2, max2, den2);
    edge_msg_kernel<<<eb, 256, 0, stream>>>(ei, E, N, al_s2, al_d2, max2, den2,
                                            h2, out2, DOUT, H2OUT, DOUT, 0, 1.0f / HEADS);

    final_kernel<<<(N + 255) / 256, 256, 0, stream>>>(out2, b2, out, N);
}
